// LSTMWithCellAttentionVector_69191923139130
// MI455X (gfx1250) — matrix operand order repaired and checked by reading, not compiled, not run on hardware
//
#include <hip/hip_runtime.h>
#include <hip/hip_bf16.h>

// Problem dims
#define BS   64
#define SEQ  512
#define IN   256
#define HS   512
#define RR   32
#define DA   128
#define NROW (BS * SEQ)        // 32768 rows of (b,t)
#define KBIG (RR * IN)         // 8192
#define NG   (4 * HS)          // 2048

#define BLSTRIDE 24            // padded per-lane stride (shorts) in swizzled B LDS
#define BSWSZ    (8 * 32 * BLSTRIDE)

typedef __bf16 v16bf __attribute__((ext_vector_type(16)));
typedef float  v8f   __attribute__((ext_vector_type(8)));

union BF16x16 { unsigned short u[16]; v16bf v; };

__device__ __forceinline__ unsigned short f2bf(float f) {
    unsigned u = __float_as_uint(f);
    return (unsigned short)((u + 0x7FFFu + ((u >> 16) & 1u)) >> 16);
}
__device__ __forceinline__ float bf2f(unsigned short s) {
    return __uint_as_float(((unsigned)s) << 16);
}
__device__ __forceinline__ float sigmoidf(float v) { return 1.0f / (1.0f + __expf(-v)); }

// ---- WMMA fragment helpers (wave32; A and B both use contiguous k = 16*(lane>>4) + h) -----------
// A (bf16 16x32) from row-major LDS [16][stride]: lane row M=lane&15;
// K = (lane>>4)*16 + h  -> one contiguous 16-short run
__device__ __forceinline__ v16bf load_a_frag(const unsigned short* base, int stride, int lane) {
    BF16x16 af;
    int m  = lane & 15;
    int ks = (lane >> 4) * 16;
#pragma unroll
    for (int h = 0; h < 16; ++h) {
        int K = ks + h;
        af.u[h] = base[m * stride + K];
    }
    return af.v;
}
// Swizzled-B LDS index for element (kr in 0..31, c = local col): the 16 values
// each lane needs are stored contiguously -> fragment load = 2x ds_load_b128.
__device__ __forceinline__ int bswz_idx(int kr, int c) {
    return (((c >> 4) << 5) + ((kr >> 4) << 4) + (c & 15)) * BLSTRIDE + (kr & 15);
}
__device__ __forceinline__ v16bf load_b_frag_sw(const unsigned short* Bsw, int tile, int lane) {
    const unsigned short* p = Bsw + ((tile << 5) + lane) * BLSTRIDE;
    BF16x16 bf;
#pragma unroll
    for (int h = 0; h < 16; ++h) bf.u[h] = p[h];
    return bf.v;
}

// ---- batched staging helpers -----------------------------------------------
// A stage: thread owns (m = tid>>2, k0 = (tid&3)*8): 8 contiguous fp32 loads
// into regs, then 4 packed ds_store_b32 of bf16 pairs.
__device__ __forceinline__ void g_load_a8(const float* xp, float xv[8]) {
#pragma unroll
    for (int q = 0; q < 8; ++q) xv[q] = xp[q];
}
__device__ __forceinline__ void s_store_a8(unsigned short* As, int m, int k0,
                                           const float xv[8], float scale) {
#pragma unroll
    for (int q = 0; q < 4; ++q) {
        unsigned lo = f2bf(scale * xv[2 * q]);
        unsigned hi = f2bf(scale * xv[2 * q + 1]);
        *(unsigned*)&As[m * 40 + k0 + 2 * q] = lo | (hi << 16);
    }
}
// B stage: thread owns a 4x4 (kr x c) sub-tile: 4 contiguous b64 global loads,
// then 4 ds_store_b64 (the swizzle makes kr the fastest axis, so 4 k-values
// for one column pack into one 64-bit LDS word).
__device__ __forceinline__ void g_load_b44(const unsigned short* g, size_t stride,
                                           unsigned short rv[4][4]) {
#pragma unroll
    for (int j = 0; j < 4; ++j)
#pragma unroll
        for (int i = 0; i < 4; ++i) rv[j][i] = g[(size_t)j * stride + i];
}
__device__ __forceinline__ void s_store_b44(unsigned short* Bsw, int kr0, int c0,
                                            const unsigned short rv[4][4]) {
#pragma unroll
    for (int i = 0; i < 4; ++i) {
        unsigned long long pk = (unsigned long long)rv[0][i]
            | ((unsigned long long)rv[1][i] << 16)
            | ((unsigned long long)rv[2][i] << 32)
            | ((unsigned long long)rv[3][i] << 48);
        *(unsigned long long*)&Bsw[bswz_idx(kr0, c0 + i)] = pk;
    }
}

// ---- Phase 0: convert weights to bf16 (W_iBarh, W_hh row-major; W1 transposed)
__global__ void __launch_bounds__(256) prep_kernel(
    const float* __restrict__ Wi, const float* __restrict__ Whh,
    const float* __restrict__ W1,
    unsigned short* __restrict__ Wi_bf, unsigned short* __restrict__ Whh_bf,
    unsigned short* __restrict__ W1T_bf) {
    size_t idx = (size_t)blockIdx.x * 256 + threadIdx.x;
    if (idx < (size_t)KBIG * NG) Wi_bf[idx] = f2bf(Wi[idx]);
    if (idx < (size_t)HS * NG)   Whh_bf[idx] = f2bf(Whh[idx]);
    if (idx < (size_t)IN * DA) {            // W1T[k][n] = W1[n][k], W1 is [DA][IN]
        int k = (int)(idx >> 7), n = (int)(idx & 127);
        W1T_bf[idx] = f2bf(W1[n * IN + k]);
    }
}

// ---- Phase 1: fused attention: softmax(tanh(x W1^T + b1) W2^T + b2) --------
__global__ void __launch_bounds__(256) attention_kernel(
    const float* __restrict__ x, const unsigned short* __restrict__ W1T,
    const float* __restrict__ b1, const float* __restrict__ W2,
    const float* __restrict__ b2, float* __restrict__ att) {
    __shared__ __align__(16) unsigned short As[64 * 40];
    __shared__ __align__(16) unsigned short Bsw[BSWSZ];
    __shared__ unsigned short H1s[64][128];
    __shared__ float atts[64][33];

    int row0 = blockIdx.x * 64;
    int tid = threadIdx.x, lane = tid & 31, wv = tid >> 5;
    int wr = wv >> 1;              // 16-row sub-block
    int wc4 = (wv & 1) * 4;        // tile index base (each tile = 16 cols)
    int am = tid >> 2, ak0 = (tid & 3) * 8;       // A staging role
    int bc0 = (tid & 31) * 4, bkr0 = (tid >> 5) * 4;  // B staging role
    v8f acc[4] = {};

    for (int kk = 0; kk < IN; kk += 32) {
        float xv[8];
        unsigned short rv[4][4];
        g_load_a8(x + (size_t)(row0 + am) * IN + kk + ak0, xv);
        g_load_b44(W1T + (size_t)(kk + bkr0) * DA + bc0, DA, rv);
        s_store_a8(As, am, ak0, xv, 1.0f);
        s_store_b44(Bsw, bkr0, bc0, rv);
        __syncthreads();
        v16bf af = load_a_frag(&As[wr * 16 * 40], 40, lane);
#pragma unroll
        for (int t4 = 0; t4 < 4; ++t4) {
            v16bf bf = load_b_frag_sw(Bsw, wc4 + t4, lane);
            acc[t4] = __builtin_amdgcn_wmma_f32_16x16x32_bf16(
                false, af, false, bf, (short)0, acc[t4], false, false);
        }
        __syncthreads();
    }
    {   // epilogue GEMM1: + b1, tanh, store bf16 to LDS
        int n = lane & 15, hi = lane >> 4;
#pragma unroll
        for (int t4 = 0; t4 < 4; ++t4) {
            int col = (wc4 + t4) * 16 + n;
#pragma unroll
            for (int v = 0; v < 8; ++v) {
                int m = wr * 16 + v + 8 * hi;
                H1s[m][col] = f2bf(tanhf(acc[t4][v] + b1[col]));
            }
        }
    }
    __syncthreads();
    {   // GEMM2: (64x128)x(128x32) scalar, 4 threads per row x 8 r each
        int row = tid >> 2, r0 = (tid & 3) * 8;
#pragma unroll
        for (int rr = 0; rr < 8; ++rr) {
            int r = r0 + rr;
            float s = b2[r];
            for (int kc = 0; kc < DA; ++kc)
                s += bf2f(H1s[row][kc]) * W2[r * DA + kc];
            atts[row][r] = s;
        }
    }
    __syncthreads();
    if (tid < 64) {  // softmax over 32 slots
        float mx = -1e30f;
        for (int r = 0; r < RR; ++r) mx = fmaxf(mx, atts[tid][r]);
        float sum = 0.f;
        for (int r = 0; r < RR; ++r) { atts[tid][r] = __expf(atts[tid][r] - mx); sum += atts[tid][r]; }
        float inv = 1.0f / sum;
        float* ap = att + (size_t)(row0 + tid) * RR;
        for (int r = 0; r < RR; ++r) ap[r] = atts[tid][r] * inv;
    }
}

// ---- Phase 2: G[bt,j] = (att_t (x) x_t) @ W_iBarh, A built on the fly ------
// Grid: (NROW/64, NG/128). K = 8192. bf16 WMMA, fp32 accumulate.
// Double-buffered LDS: next K-step's global loads overlap current WMMAs.
__global__ void __launch_bounds__(256) gemm_xbar_kernel(
    const float* __restrict__ x, const float* __restrict__ att,
    const unsigned short* __restrict__ Wi, float* __restrict__ G) {
    __shared__ __align__(16) unsigned short As[2][64 * 40];
    __shared__ __align__(16) unsigned short Bsw[2][BSWSZ];
    int row0 = blockIdx.x * 64;
    int col0 = blockIdx.y * 128;
    int tid = threadIdx.x, lane = tid & 31, wv = tid >> 5;
    int wr = wv >> 1, wc4 = (wv & 1) * 4;
    int am = tid >> 2, ak0 = (tid & 3) * 8;
    int bc0 = (tid & 31) * 4, bkr0 = (tid >> 5) * 4;
    int arow = row0 + am;
    const float* xrow = x + (size_t)arow * IN;
    const float* arow_att = att + (size_t)arow * RR;
    v8f acc[4] = {};

    float xv[8];
    float av;
    unsigned short rv[4][4];
    // prologue: stage K-step 0
    g_load_a8(xrow + ak0, xv);
    av = arow_att[0];
    g_load_b44(Wi + (size_t)bkr0 * NG + col0 + bc0, NG, rv);
    s_store_a8(As[0], am, ak0, xv, av);
    s_store_b44(Bsw[0], bkr0, bc0, rv);
    __syncthreads();

    int buf = 0;
    for (int kk = 0; kk < KBIG; kk += 32) {
        int nkk = kk + 32;
        bool have_next = (nkk < KBIG);
        if (have_next) {   // issue next-step global loads early
            g_load_a8(xrow + (nkk & 255) + ak0, xv);
            av = arow_att[nkk >> 8];
            g_load_b44(Wi + (size_t)(nkk + bkr0) * NG + col0 + bc0, NG, rv);
        }
        v16bf af = load_a_frag(&As[buf][wr * 16 * 40], 40, lane);
#pragma unroll
        for (int t4 = 0; t4 < 4; ++t4) {
            v16bf bf = load_b_frag_sw(Bsw[buf], wc4 + t4, lane);
            acc[t4] = __builtin_amdgcn_wmma_f32_16x16x32_bf16(
                false, af, false, bf, (short)0, acc[t4], false, false);
        }
        if (have_next) {
            s_store_a8(As[buf ^ 1], am, ak0, xv, av);
            s_store_b44(Bsw[buf ^ 1], bkr0, bc0, rv);
        }
        __syncthreads();
        buf ^= 1;
    }
    int n = lane & 15, hi = lane >> 4;
#pragma unroll
    for (int t4 = 0; t4 < 4; ++t4) {
        int c = col0 + (wc4 + t4) * 16 + n;
#pragma unroll
        for (int v = 0; v < 8; ++v) {
            int m = row0 + wr * 16 + v + 8 * hi;
            G[(size_t)m * NG + c] = acc[t4][v];
        }
    }
}

// ---- Phase 3: norms[b][t] = ||M_t||, M held in LDS (32KB), one WG per batch
__global__ void __launch_bounds__(256) norm_kernel(
    const float* __restrict__ x, const float* __restrict__ att,
    float* __restrict__ norms) {
    __shared__ float M[RR][IN + 1];
    __shared__ float xs[IN];
    __shared__ float red[8];
    int b = blockIdx.x, tid = threadIdx.x;
    int r = tid >> 3, i0 = (tid & 7) * 32;
#pragma unroll
    for (int q = 0; q < 32; ++q) M[r][i0 + q] = 0.f;
    for (int t = 0; t < SEQ; ++t) {
        __syncthreads();
        xs[tid] = x[((size_t)b * SEQ + t) * IN + tid];
        __syncthreads();
        float a = att[((size_t)b * SEQ + t) * RR + r];
        float ss = 0.f;
#pragma unroll
        for (int q = 0; q < 32; ++q) {
            float v = M[r][i0 + q] + a * xs[i0 + q];
            M[r][i0 + q] = v;
            ss += v * v;
        }
        for (int off = 16; off > 0; off >>= 1) ss += __shfl_down(ss, off, 32);
        if ((tid & 31) == 0) red[tid >> 5] = ss;
        __syncthreads();
        if (tid == 0) {
            float tot = 0.f;
            for (int w = 0; w < 8; ++w) tot += red[w];
            norms[(size_t)b * SEQ + t] = sqrtf(tot);
        }
    }
}

// ---- Phase 4: per-timestep recurrence. Each WG owns a 32-wide hs block and
// computes all 4 gate strips (i/f/g/o) so the cell update fuses in-register.
__global__ void __launch_bounds__(256) lstm_step_kernel(
    const unsigned short* __restrict__ Whh, const float* __restrict__ G,
    const float* __restrict__ norms, const float* __restrict__ bias,
    float* __restrict__ P, float* __restrict__ c_ws,
    float* __restrict__ hseq, float* __restrict__ hT, float* __restrict__ cT,
    int t) {
    __shared__ __align__(16) unsigned short hA[64 * 40];
    __shared__ __align__(16) unsigned short Bsw[BSWSZ];
    int hs0 = blockIdx.x * 32;
    int tid = threadIdx.x, lane = tid & 31, wv = tid >> 5;
    int wr = wv >> 1;            // 16-batch-row block
    int ch = (wv & 1) * 16;      // 16-col half within hs block
    int am = tid >> 2, ak0 = (tid & 3) * 8;
    int bc0 = (tid & 31) * 4, bkr0 = (tid >> 5) * 4;
    int bgate = bc0 >> 5;        // gate strip for this thread's column run
    int bcol = bgate * HS + hs0 + (bc0 & 31);
    v8f acc[4] = {};

    for (int kk = 0; kk < HS; kk += 32) {
        float xv[8];
        unsigned short rv[4][4];
        if (t > 0)
            g_load_a8(hseq + ((size_t)am * SEQ + (t - 1)) * HS + kk + ak0, xv);
        else {
#pragma unroll
            for (int q = 0; q < 8; ++q) xv[q] = 0.f;
        }
        g_load_b44(Whh + (size_t)(kk + bkr0) * NG + bcol, NG, rv);
        s_store_a8(hA, am, ak0, xv, 1.0f);
        s_store_b44(Bsw, bkr0, bc0, rv);
        __syncthreads();
        v16bf af = load_a_frag(&hA[wr * 16 * 40], 40, lane);
#pragma unroll
        for (int g = 0; g < 4; ++g) {
            v16bf bf = load_b_frag_sw(Bsw, ((g * 32 + ch) >> 4), lane);
            acc[g] = __builtin_amdgcn_wmma_f32_16x16x32_bf16(
                false, af, false, bf, (short)0, acc[g], false, false);
        }
        __syncthreads();
    }
    // epilogue: running P, normalization, bias, LSTM cell
    int n = lane & 15, hi = lane >> 4;
    int hs = hs0 + ch + n;
#pragma unroll
    for (int v = 0; v < 8; ++v) {
        int b = wr * 16 + v + 8 * hi;    // batch index 0..63
        float invn = 1.0f / fmaxf(norms[(size_t)b * SEQ + t], 1e-12f);
        float gv[4];
#pragma unroll
        for (int g = 0; g < 4; ++g) {
            int j = g * HS + hs;
            float Pv = (t == 0 ? 0.f : P[(size_t)b * NG + j]) +
                       G[((size_t)b * SEQ + t) * NG + j];
            P[(size_t)b * NG + j] = Pv;
            gv[g] = acc[g][v] + Pv * invn + bias[j];
        }
        float ig = sigmoidf(gv[0]), fg = sigmoidf(gv[1]);
        float gg = tanhf(gv[2]),   og = sigmoidf(gv[3]);
        float cold = (t == 0) ? 0.f : c_ws[(size_t)b * HS + hs];
        float cnew = fg * cold + ig * gg;
        float h = og * tanhf(cnew);
        c_ws[(size_t)b * HS + hs] = cnew;
        hseq[((size_t)b * SEQ + t) * HS + hs] = h;
        if (t == SEQ - 1) { hT[(size_t)b * HS + hs] = h; cT[(size_t)b * HS + hs] = cnew; }
    }
}

static inline size_t alignup(size_t v) { return (v + 255) & ~(size_t)255; }

extern "C" void kernel_launch(void* const* d_in, const int* in_sizes, int n_in,
                              void* d_out, int out_size, void* d_ws, size_t ws_size,
                              hipStream_t stream) {
    const float* x    = (const float*)d_in[0];
    const float* W1   = (const float*)d_in[1];
    const float* b1   = (const float*)d_in[2];
    const float* W2   = (const float*)d_in[3];
    const float* b2   = (const float*)d_in[4];
    const float* Wi   = (const float*)d_in[5];
    const float* Whh  = (const float*)d_in[6];
    const float* bias = (const float*)d_in[7];

    float* out  = (float*)d_out;
    float* hseq = out;                                   // (64,512,512)
    float* hT   = out + (size_t)BS * SEQ * HS;           // (64,512)
    float* cT   = hT + (size_t)BS * HS;                  // (64,512)

    // Workspace layout (~295 MB total; G dominates)
    char* ws = (char*)d_ws;
    unsigned short* Wi_bf  = (unsigned short*)ws; ws += alignup((size_t)KBIG * NG * 2);
    unsigned short* Whh_bf = (unsigned short*)ws; ws += alignup((size_t)HS * NG * 2);
    unsigned short* W1T_bf = (unsigned short*)ws; ws += alignup((size_t)IN * DA * 2);
    float* att   = (float*)ws; ws += alignup((size_t)NROW * RR * 4);
    float* norms = (float*)ws; ws += alignup((size_t)NROW * 4);
    float* P     = (float*)ws; ws += alignup((size_t)BS * NG * 4);
    float* c_ws  = (float*)ws; ws += alignup((size_t)BS * HS * 4);
    float* G     = (float*)ws; ws += alignup((size_t)NROW * NG * 4);
    (void)in_sizes; (void)n_in; (void)out_size; (void)ws_size;

    // Phase 0: bf16 weight conversion
    prep_kernel<<<(KBIG * NG) / 256, 256, 0, stream>>>(Wi, Whh, W1, Wi_bf, Whh_bf, W1T_bf);
    // Phase 1: attention weights for all (b,t)
    attention_kernel<<<NROW / 64, 256, 0, stream>>>(x, W1T_bf, b1, W2, b2, att);
    // Phase 2: the big GEMM, parallel over all timesteps
    dim3 g2(NROW / 64, NG / 128);
    gemm_xbar_kernel<<<g2, 256, 0, stream>>>(x, att, Wi_bf, G);
    // Phase 3: norms of M_t
    norm_kernel<<<BS, 256, 0, stream>>>(x, att, norms);
    // Phase 4: sequential recurrence, 512 lightweight WMMA steps
    for (int t = 0; t < SEQ; ++t)
        lstm_step_kernel<<<NG / 128, 256, 0, stream>>>(Whh_bf, G, norms, bias,
                                                       P, c_ws, hseq, hT, cT, t);
}
